// MoE_87582973100424
// MI455X (gfx1250) — compile-verified, matrix operand order checked by reading, not run on hardware
//
#include <hip/hip_runtime.h>
#include <hip/hip_bf16.h>
#include <math.h>

#define B_ROWS  8192
#define IN_DIM  1024
#define HID_DIM 4096
#define OUT_DIM 1024
#define NEXP    8
#define LN_EPS  1e-5f

typedef __attribute__((ext_vector_type(16))) __bf16 v16bf;
typedef __attribute__((ext_vector_type(8)))  float  v8f;

union FragU {
    v16bf v;
    unsigned int u[8];
};

__device__ __forceinline__ unsigned short f32_to_bf16(float f) {
    unsigned int x = __builtin_bit_cast(unsigned int, f);
    unsigned int r = x + 0x7FFFu + ((x >> 16) & 1u);   // round-to-nearest-even
    return (unsigned short)(r >> 16);
}

__device__ __forceinline__ float bf16_to_f32(unsigned short h) {
    unsigned int x = ((unsigned int)h) << 16;
    return __builtin_bit_cast(float, x);
}

// Load a 16x32 bf16 fragment (A-layout per ISA 7.12.2; B uses the mirrored
// layout with row_sel = N column index into a K-contiguous (transposed) matrix).
// K packing per lane: vgpr v holds K = k0 + grp*8 + (v&3)*2 + (v>>2)*16 (pair).
__device__ __forceinline__ v16bf load_frag(const unsigned short* __restrict__ base,
                                           int stride, int row_sel, int k0, int grp) {
    FragU f;
    const unsigned short* p = base + (size_t)row_sel * stride + k0 + grp * 8;
#pragma unroll
    for (int v = 0; v < 8; ++v) {
        int ko = (v & 3) * 2 + (v >> 2) * 16;
        f.u[v] = *(const unsigned int*)(p + ko);
    }
    return f.v;
}

__device__ __forceinline__ float gelu_exact(float v) {
    return 0.5f * v * (1.0f + erff(v * 0.70710678118654752f));
}

// ---------------------------------------------------------------------------
// One-shot: W [E][K][N] f32  ->  Wt [E][N][K] bf16  (K-contiguous rows so WMMA
// B-fragments are dword loads). 32x32 tiles through LDS, coalesced both ways.
// ---------------------------------------------------------------------------
__global__ void transpose_to_bf16(const float* __restrict__ w,
                                  unsigned short* __restrict__ wt,
                                  int K, int N) {
    __shared__ float tile[32][33];
    const int e  = blockIdx.z;
    const int k0 = blockIdx.y * 32;
    const int n0 = blockIdx.x * 32;
    const float* we = w + (size_t)e * K * N;
    unsigned short* wte = wt + (size_t)e * K * N;
    const int tx = threadIdx.x & 31;
    const int ty = threadIdx.x >> 5;          // 8 rows per step
#pragma unroll
    for (int i = 0; i < 32; i += 8)
        tile[ty + i][tx] = we[(size_t)(k0 + ty + i) * N + (n0 + tx)];
    __syncthreads();
#pragma unroll
    for (int i = 0; i < 32; i += 8)
        wte[(size_t)(n0 + ty + i) * K + (k0 + tx)] = f32_to_bf16(tile[tx][ty + i]);
}

// ---------------------------------------------------------------------------
// Fused MoE: per WG = 16 token rows, loop over 8 experts.
// GEMM1 (bf16 WMMA) -> bias -> LN -> GELU (h in LDS as bf16)
// GEMM2 (bf16 WMMA) -> bias -> LN -> GELU -> weighted accumulate (LDS) -> out
// ---------------------------------------------------------------------------
__global__ __launch_bounds__(256, 2)
void moe_fused(const float* __restrict__ x,
               const float* __restrict__ wgt,      // [B, E]
               const float* __restrict__ b1,
               const float* __restrict__ g1,
               const float* __restrict__ be1,
               const float* __restrict__ b2,
               const float* __restrict__ g2,
               const float* __restrict__ be2,
               const unsigned short* __restrict__ w1t,  // [E][HID][IN] bf16
               const unsigned short* __restrict__ w2t,  // [E][OUT][HID] bf16
               float* __restrict__ out) {
    extern __shared__ char smem[];
    unsigned short* lds_x  = (unsigned short*)smem;                        // 16*1024 bf16
    unsigned short* lds_h  = (unsigned short*)(smem + 16 * IN_DIM * 2);    // 16*4096 bf16
    float* lds_out = (float*)(smem + 16 * IN_DIM * 2 + 16 * HID_DIM * 2);  // 16*1024 f32
    float* lds_red = (float*)(smem + 16 * IN_DIM * 2 + 16 * HID_DIM * 2
                                   + 16 * OUT_DIM * 4);
    // lds_red: [0..15]=sum  [16..31]=sumsq  [32..47]=mu  [48..63]=rsigma  [64..79]=gate wt

    const int tid  = threadIdx.x;
    const int lane = tid & 31;
    const int wave = tid >> 5;        // 0..7
    const int grp  = lane >> 4;       // 0/1 -> K lane-group
    const int ln   = lane & 15;       // M (A/C) or N (B/C) index
    const int m0   = blockIdx.x * 16;

    // ---- init output accumulator + stage x tile (f32 -> bf16 LDS)
    for (int i = tid; i < 16 * OUT_DIM; i += 256) lds_out[i] = 0.f;
    for (int i = tid; i < 16 * IN_DIM; i += 256) {
        int r = i >> 10, c = i & (IN_DIM - 1);
        lds_x[i] = f32_to_bf16(x[(size_t)(m0 + r) * IN_DIM + c]);
    }
    __syncthreads();

    const v8f vzero = {0.f, 0.f, 0.f, 0.f, 0.f, 0.f, 0.f, 0.f};

    for (int e = 0; e < NEXP; ++e) {
        const unsigned short* w1e = w1t + (size_t)e * HID_DIM * IN_DIM;
        const unsigned short* w2e = w2t + (size_t)e * OUT_DIM * HID_DIM;

        if (tid < 32) lds_red[tid] = 0.f;
        if (tid < 16) lds_red[64 + tid] = wgt[(size_t)(m0 + tid) * NEXP + e];
        __syncthreads();

        // ================= GEMM1: h[16, 4096] = x_tile @ W1[e] ==============
        float psum[8], psumsq[8];
#pragma unroll
        for (int r = 0; r < 8; ++r) { psum[r] = 0.f; psumsq[r] = 0.f; }

        for (int pass = 0; pass < 2; ++pass) {
            v8f acc[16];
#pragma unroll
            for (int t = 0; t < 16; ++t) acc[t] = vzero;
            const int nbase = wave * 512 + pass * 256;

            for (int k0 = 0; k0 < IN_DIM; k0 += 32) {
                v16bf a = load_frag(lds_x, IN_DIM, ln, k0, grp);
                __builtin_prefetch(w1e + (size_t)(nbase + ln) * IN_DIM + k0 + 64, 0, 3);
                // software-pipelined B fragments: load t+1 behind wmma(t)
                v16bf bcur = load_frag(w1e, IN_DIM, nbase + ln, k0, grp);
#pragma unroll
                for (int t = 0; t < 16; ++t) {
                    v16bf bnext = bcur;
                    if (t < 15)
                        bnext = load_frag(w1e, IN_DIM, nbase + (t + 1) * 16 + ln, k0, grp);
                    acc[t] = __builtin_amdgcn_wmma_f32_16x16x32_bf16(
                        false, a, false, bcur, (short)0, acc[t], false, false);
                    bcur = bnext;
                }
            }
            // epilogue: + bias, LN partial stats, store bf16 h to LDS
#pragma unroll
            for (int t = 0; t < 16; ++t) {
                const int n = nbase + t * 16 + ln;
                const float bias = b1[e * HID_DIM + n];
#pragma unroll
                for (int r = 0; r < 8; ++r) {
                    float v = acc[t][r] + bias;
                    psum[r]   += v;
                    psumsq[r] += v * v;
                    const int m = r + 8 * grp;
                    lds_h[m * HID_DIM + n] = f32_to_bf16(v);
                }
            }
        }
#pragma unroll
        for (int r = 0; r < 8; ++r) {
            const int m = r + 8 * grp;
            atomicAdd(&lds_red[m], psum[r]);
            atomicAdd(&lds_red[16 + m], psumsq[r]);
        }
        __syncthreads();
        if (tid < 16) {
            float mu  = lds_red[tid] * (1.0f / HID_DIM);
            float var = lds_red[16 + tid] * (1.0f / HID_DIM) - mu * mu;
            lds_red[32 + tid] = mu;
            lds_red[48 + tid] = rsqrtf(var + LN_EPS);
        }
        __syncthreads();
        // in-place LN + GELU over bf16 h tile
        for (int i = tid; i < 16 * HID_DIM; i += 256) {
            const int m = i >> 12, n = i & (HID_DIM - 1);
            float v = bf16_to_f32(lds_h[i]);
            v = (v - lds_red[32 + m]) * lds_red[48 + m] * g1[e * HID_DIM + n]
                + be1[e * HID_DIM + n];
            lds_h[i] = f32_to_bf16(gelu_exact(v));
        }
        __syncthreads();
        if (tid < 32) lds_red[tid] = 0.f;
        __syncthreads();

        // ================= GEMM2: y[16, 1024] = h @ W2[e] ===================
        v8f acc2[8];
#pragma unroll
        for (int t = 0; t < 8; ++t) acc2[t] = vzero;
        const int nbase2 = wave * 128;

        for (int k0 = 0; k0 < HID_DIM; k0 += 32) {
            v16bf a = load_frag(lds_h, HID_DIM, ln, k0, grp);
            __builtin_prefetch(w2e + (size_t)(nbase2 + ln) * HID_DIM + k0 + 64, 0, 3);
            v16bf bcur = load_frag(w2e, HID_DIM, nbase2 + ln, k0, grp);
#pragma unroll
            for (int t = 0; t < 8; ++t) {
                v16bf bnext = bcur;
                if (t < 7)
                    bnext = load_frag(w2e, HID_DIM, nbase2 + (t + 1) * 16 + ln, k0, grp);
                acc2[t] = __builtin_amdgcn_wmma_f32_16x16x32_bf16(
                    false, a, false, bcur, (short)0, acc2[t], false, false);
                bcur = bnext;
            }
        }

        float p2[8], p2sq[8];
#pragma unroll
        for (int r = 0; r < 8; ++r) { p2[r] = 0.f; p2sq[r] = 0.f; }
#pragma unroll
        for (int t = 0; t < 8; ++t) {
            const int n = nbase2 + t * 16 + ln;
            const float bias = b2[e * OUT_DIM + n];
#pragma unroll
            for (int r = 0; r < 8; ++r) {
                float v = acc2[t][r] + bias;
                acc2[t][r] = v;
                p2[r]   += v;
                p2sq[r] += v * v;
            }
        }
#pragma unroll
        for (int r = 0; r < 8; ++r) {
            const int m = r + 8 * grp;
            atomicAdd(&lds_red[m], p2[r]);
            atomicAdd(&lds_red[16 + m], p2sq[r]);
        }
        __syncthreads();
        if (tid < 16) {
            float mu  = lds_red[tid] * (1.0f / OUT_DIM);
            float var = lds_red[16 + tid] * (1.0f / OUT_DIM) - mu * mu;
            lds_red[32 + tid] = mu;
            lds_red[48 + tid] = rsqrtf(var + LN_EPS);
        }
        __syncthreads();
        // LN + GELU + gated accumulate into LDS output tile (each (m,n) is
        // owned by exactly one lane -> race-free read-modify-write)
#pragma unroll
        for (int r = 0; r < 8; ++r) {
            const int m = r + 8 * grp;
            const float mu = lds_red[32 + m];
            const float ri = lds_red[48 + m];
            const float gw = lds_red[64 + m];
#pragma unroll
            for (int t = 0; t < 8; ++t) {
                const int n = nbase2 + t * 16 + ln;
                float v = (acc2[t][r] - mu) * ri * g2[e * OUT_DIM + n]
                          + be2[e * OUT_DIM + n];
                lds_out[m * OUT_DIM + n] += gw * gelu_exact(v);
            }
        }
        __syncthreads();   // protect lds_h / lds_red before next expert
    }

    // ---- final store: coalesced sweep of the accumulated output tile
    for (int i = tid; i < 16 * OUT_DIM; i += 256) {
        const int m = i >> 10, n = i & (OUT_DIM - 1);
        out[(size_t)(m0 + m) * OUT_DIM + n] = lds_out[i];
    }
}

extern "C" void kernel_launch(void* const* d_in, const int* in_sizes, int n_in,
                              void* d_out, int out_size, void* d_ws, size_t ws_size,
                              hipStream_t stream) {
    const float* x   = (const float*)d_in[0];
    const float* wgt = (const float*)d_in[1];
    const float* W1  = (const float*)d_in[2];
    const float* b1  = (const float*)d_in[3];
    const float* g1  = (const float*)d_in[4];
    const float* be1 = (const float*)d_in[5];
    const float* W2  = (const float*)d_in[6];
    const float* b2  = (const float*)d_in[7];
    const float* g2  = (const float*)d_in[8];
    const float* be2 = (const float*)d_in[9];
    float* out = (float*)d_out;

    // workspace: bf16 transposed weights (W1t: 64 MB, W2t: 64 MB) -> L2-resident
    unsigned short* w1t = (unsigned short*)d_ws;
    unsigned short* w2t = w1t + (size_t)NEXP * IN_DIM * HID_DIM;

    transpose_to_bf16<<<dim3(HID_DIM / 32, IN_DIM / 32, NEXP), dim3(256), 0, stream>>>(
        W1, w1t, IN_DIM, HID_DIM);
    transpose_to_bf16<<<dim3(OUT_DIM / 32, HID_DIM / 32, NEXP), dim3(256), 0, stream>>>(
        W2, w2t, HID_DIM, OUT_DIM);

    const size_t lds_bytes = (size_t)16 * IN_DIM * 2   // x tile bf16
                           + (size_t)16 * HID_DIM * 2  // h tile bf16
                           + (size_t)16 * OUT_DIM * 4  // output accumulator f32
                           + 80 * sizeof(float);       // reduction scratch
    moe_fused<<<dim3(B_ROWS / 16), dim3(256), lds_bytes, stream>>>(
        x, wgt, b1, g1, be1, b2, g2, be2, w1t, w2t, out);
}
